// KeyPointMLPDeocder_39625368273066
// MI455X (gfx1250) — compile-verified, matrix operand order checked by reading, not run on hardware
//
#include <hip/hip_runtime.h>
#include <hip/hip_bf16.h>
#include <math.h>

// ---------------------------------------------------------------------------
// Problem constants (from reference)
// ---------------------------------------------------------------------------
#define BB   2048
#define SS   134
#define EE   256
#define HH   1024
#define TT   5          // NUM_KP
#define KK   6
#define MM   (BB * TT)  // 10240 rows
#define LN_EPS 1e-5f
#define NTILE_M (MM / 16)   // 640 row tiles
#define NKC     (EE / 32)   // 8 K-chunks for GEMM1

// d_out flat layout (return order): kp_loss, logits, scores, selected, loss_per_kp
#define OFF_LOGITS 1
#define OFF_SCORES (OFF_LOGITS + (size_t)MM * 12)   // 122881
#define OFF_SEL    (OFF_SCORES + (size_t)MM * 6)    // 184321
#define OFF_LPP    (OFF_SEL    + (size_t)MM * 2)    // 204801

// ---------------------------------------------------------------------------
// Workspace layout (bytes). Fragment = 32 lanes x (16 bf16 hi + 16 bf16 lo)
//   = 1024 bf16 = 2048 B, stored hi-block (512 elem) then lo-block (512 elem).
// A pack:  [tileM(640)][kc(8)]          -> 5120 frags = 10,485,760 B (shared)
// B pack:  [ntileN(64)][kc(8)] per dec  ->  512 frags =  1,048,576 B each
// reduce scratch: 2 * T * B floats      ->       81,920 B
// total ~12.7 MB
// ---------------------------------------------------------------------------
#define FRAG_ELEMS 1024
#define WS_APACK_OFF 0
#define WS_APACK_BYTES ((size_t)NTILE_M * NKC * FRAG_ELEMS * 2)
#define WS_BPACK_LD_OFF (WS_APACK_OFF + WS_APACK_BYTES)
#define WS_BPACK_BYTES ((size_t)64 * NKC * FRAG_ELEMS * 2)
#define WS_BPACK_SD_OFF (WS_BPACK_LD_OFF + WS_BPACK_BYTES)
#define WS_RED_OFF (WS_BPACK_SD_OFF + WS_BPACK_BYTES)

// LDS partition (floats); strides padded to multiples of 4 floats (16B) so
// float4 LDS loads are aligned, and to break bank conflicts.
#define AS_STRIDE  260
#define MLP_STRIDE 1028
#define AS_OFF   0
#define MLP_OFF  (16 * AS_STRIDE)                     // 4160
#define W2T_OFF  (MLP_OFF + 16 * MLP_STRIDE)          // 20608
#define SUM_OFF  (W2T_OFF + 1280 * 12)                // 35968 (sized for C=12)
#define SQ_OFF   (SUM_OFF + 16)
#define SMEM_FLOATS (SQ_OFF + 16)                     // 36000 floats = 144000 B

typedef __attribute__((ext_vector_type(16))) __bf16 v16bf;
typedef __attribute__((ext_vector_type(8)))  float  v8f;

// K index inside a 32-wide chunk for the ISA 16-bit A/B fragment layout:
// lane half (lane>>4) selects K sub-block 0/8; halves h=0..15 map as below.
__device__ __forceinline__ int kfrag(int lane, int h) {
    return ((h & 8) << 1) + (((lane >> 4) & 1) << 3) + (h & 7);
}

// ---------------------------------------------------------------------------
// Pack the sliced activation x[M,256] into bf16 hi/lo WMMA fragments (once,
// shared by both decoders). One thread per fragment-lane.
// ---------------------------------------------------------------------------
__global__ void __launch_bounds__(256)
kp_pack_a_kernel(const float* __restrict__ hidden,
                 const int* __restrict__ ctx,
                 __bf16* __restrict__ apack)
{
    int id = blockIdx.x * blockDim.x + threadIdx.x;   // [0, 640*8*32)
    if (id >= NTILE_M * NKC * 32) return;
    int lane  = id & 31;
    int kc    = (id >> 5) & (NKC - 1);
    int tileM = id >> 8;

    int kp_start = ctx[0] - 1;
    int m  = lane & 15;
    int gr = tileM * 16 + m;
    int b  = gr / TT;
    int t  = gr - b * TT;
    const float* row = hidden + ((size_t)(b * SS + kp_start + t)) * EE + kc * 32;

    size_t frag = (size_t)tileM * NKC + kc;
    __bf16* hi = apack + frag * FRAG_ELEMS + lane * 16;
    __bf16* lo = hi + 512;
    #pragma unroll
    for (int h = 0; h < 16; ++h) {
        float x = row[kfrag(lane, h)];
        __bf16 xh = (__bf16)x;
        hi[h] = xh;
        lo[h] = (__bf16)(x - (float)xh);
    }
}

// ---------------------------------------------------------------------------
// Pack one decoder's w1[256,1024] into bf16 hi/lo WMMA B-fragments.
// ---------------------------------------------------------------------------
__global__ void __launch_bounds__(256)
kp_pack_w1_kernel(const float* __restrict__ w1,
                  __bf16* __restrict__ bpack)
{
    int id = blockIdx.x * blockDim.x + threadIdx.x;   // [0, 64*8*32)
    if (id >= 64 * NKC * 32) return;
    int lane  = id & 31;
    int kc    = (id >> 5) & (NKC - 1);
    int ntile = id >> 8;

    int n = ntile * 16 + (lane & 15);
    const float* col = w1 + n + (size_t)(kc * 32) * HH;

    size_t frag = (size_t)ntile * NKC + kc;
    __bf16* hi = bpack + frag * FRAG_ELEMS + lane * 16;
    __bf16* lo = hi + 512;
    #pragma unroll
    for (int h = 0; h < 16; ++h) {
        float x = col[(size_t)kfrag(lane, h) * HH];
        __bf16 xh = (__bf16)x;
        hi[h] = xh;
        lo[h] = (__bf16)(x - (float)xh);
    }
}

// ---------------------------------------------------------------------------
// Fused decoder:  out = cat([x, relu(LN(x@w1+b1))]) @ w2 + b2
// grid.x = M/16; block = 256 threads = 8 waves (wave32).
// Workgroup owns 16 rows x full H; wave w owns columns [w*128, w*128+128).
// GEMM1: prepacked bf16 hi/lo fragments, 3 WMMAs per tile (hi*hi+hi*lo+lo*hi)
// on v_wmma_f32_16x16x32_bf16 -> near-fp32 fidelity at 3/8 bf16 peak.
// ---------------------------------------------------------------------------
__global__ void __launch_bounds__(256)
kp_decoder_kernel(const float* __restrict__ hidden,
                  const __bf16* __restrict__ apack,
                  const __bf16* __restrict__ bpack,
                  const float* __restrict__ b1,
                  const float* __restrict__ gamma,
                  const float* __restrict__ beta,
                  const float* __restrict__ w2,
                  const float* __restrict__ b2,
                  float* __restrict__ out,
                  const int* __restrict__ ctx,
                  int C)             // C = 12 (ld) or 6 (sd)
{
    extern __shared__ float smem[];
    float* As   = smem + AS_OFF;     // [16][260] f32 x tile (for GEMM2)
    float* Mlp  = smem + MLP_OFF;    // [16][1028] f32 post-LN/ReLU
    float* W2t  = smem + W2T_OFF;    // [C][1280] transposed w2
    float* Lsum = smem + SUM_OFF;    // [16] -> mu
    float* Lsq  = smem + SQ_OFF;     // [16] -> rstd

    const int tid  = threadIdx.x;
    const int lane = tid & 31;
    const int wave = tid >> 5;
    const int kp_start = ctx[0] - 1;

    // ---- Stage x tile (f32) for GEMM2; stage transposed w2 -----------------
    #pragma unroll 4
    for (int r = 0; r < 16; ++r) {
        int gr = blockIdx.x * 16 + r;
        int b  = gr / TT;
        int t  = gr - b * TT;
        As[r * AS_STRIDE + tid] =
            hidden[((size_t)(b * SS + kp_start + t)) * EE + tid];
    }
    for (int i = tid; i < (EE + HH) * C; i += 256) {
        int k = i / C;
        int c = i - k * C;
        W2t[c * 1280 + k] = w2[i];
    }
    if (tid < 32) smem[SUM_OFF + tid] = 0.0f;
    __syncthreads();

    // ---- GEMM1 via WMMA on prepacked fragments -----------------------------
    v8f acc[8];
    #pragma unroll
    for (int nt = 0; nt < 8; ++nt)
        #pragma unroll
        for (int j = 0; j < 8; ++j) acc[nt][j] = 0.0f;

    const __bf16* ablk = apack + ((size_t)blockIdx.x * NKC) * FRAG_ELEMS + lane * 16;

    #pragma unroll 2
    for (int kc = 0; kc < NKC; ++kc) {
        const __bf16* ap = ablk + (size_t)kc * FRAG_ELEMS;
        v16bf ahi = *(const v16bf*)ap;
        v16bf alo = *(const v16bf*)(ap + 512);
        #pragma unroll
        for (int nt = 0; nt < 8; ++nt) {
            const __bf16* bp = bpack
                + ((size_t)((wave * 8 + nt) * NKC + kc)) * FRAG_ELEMS + lane * 16;
            v16bf bhi = *(const v16bf*)bp;
            v16bf blo = *(const v16bf*)(bp + 512);
            acc[nt] = __builtin_amdgcn_wmma_f32_16x16x32_bf16(
                          false, ahi, false, bhi, (short)0, acc[nt], false, false);
            acc[nt] = __builtin_amdgcn_wmma_f32_16x16x32_bf16(
                          false, ahi, false, blo, (short)0, acc[nt], false, false);
            acc[nt] = __builtin_amdgcn_wmma_f32_16x16x32_bf16(
                          false, alo, false, bhi, (short)0, acc[nt], false, false);
        }
    }

    // ---- Bias + per-row sum / sumsq (LayerNorm statistics) -----------------
    const int ncol  = lane & 15;
    const int mbase = (lane >> 4) << 3;       // C/D row: m = mbase + j
    float psum[8], psq[8];
    #pragma unroll
    for (int j = 0; j < 8; ++j) { psum[j] = 0.0f; psq[j] = 0.0f; }

    #pragma unroll
    for (int nt = 0; nt < 8; ++nt) {
        int n = wave * 128 + nt * 16 + ncol;
        float bias = b1[n];
        #pragma unroll
        for (int j = 0; j < 8; ++j) {
            float v = acc[nt][j] + bias;
            acc[nt][j] = v;
            psum[j] += v;
            psq[j]  += v * v;
        }
    }
    #pragma unroll
    for (int j = 0; j < 8; ++j) {
        atomicAdd(&Lsum[mbase + j], psum[j]);   // ds_add_f32
        atomicAdd(&Lsq[mbase + j],  psq[j]);
    }
    __syncthreads();

    if (tid < 16) {
        float mu  = Lsum[tid] * (1.0f / HH);
        float var = Lsq[tid] * (1.0f / HH) - mu * mu;
        Lsum[tid] = mu;
        Lsq[tid]  = rsqrtf(var + LN_EPS);
    }
    __syncthreads();

    // ---- Normalize + ReLU, park MLP activations in LDS ---------------------
    #pragma unroll
    for (int nt = 0; nt < 8; ++nt) {
        int n = wave * 128 + nt * 16 + ncol;
        float g  = gamma[n];
        float be = beta[n];
        #pragma unroll
        for (int j = 0; j < 8; ++j) {
            int m = mbase + j;
            float v = (acc[nt][j] - Lsum[m]) * Lsq[m] * g + be;
            Mlp[m * MLP_STRIDE + n] = fmaxf(v, 0.0f);
        }
    }
    __syncthreads();

    // ---- GEMM2: out[m][c] = b2[c] + x . w2t[c][0:256] + mlp . w2t[c][256:] -
    if (tid < 16 * C) {
        int m = tid / C;
        int c = tid - m * C;
        float s = b2[c];
        const float* xr = &As[m * AS_STRIDE];
        const float* wt = &W2t[c * 1280];
        #pragma unroll 4
        for (int k = 0; k < EE; k += 4) {
            float4 a = *(const float4*)(xr + k);
            float4 w = *(const float4*)(wt + k);
            s += a.x * w.x + a.y * w.y + a.z * w.z + a.w * w.w;
        }
        const float* mr  = &Mlp[m * MLP_STRIDE];
        const float* wt2 = wt + EE;
        #pragma unroll 4
        for (int n = 0; n < HH; n += 4) {
            float4 a = *(const float4*)(mr + n);
            float4 w = *(const float4*)(wt2 + n);
            s += a.x * w.x + a.y * w.y + a.z * w.z + a.w * w.w;
        }
        out[(size_t)(blockIdx.x * 16 + m) * C + c] = s;
    }
}

// ---------------------------------------------------------------------------
// Per-(b,t) loss pieces. Deterministic: writes per-element partials.
// ---------------------------------------------------------------------------
__global__ void __launch_bounds__(256)
kp_loss_kernel(const float* __restrict__ logits,   // [M,12]
               const float* __restrict__ scores,   // [M,6]
               const float* __restrict__ fkp,      // [B,5,2]
               float* __restrict__ selected,       // [M,2]
               float* __restrict__ minl_ws,        // [T][B]
               float* __restrict__ ce_ws)          // [T][B]
{
    int r = blockIdx.x * blockDim.x + threadIdx.x;
    if (r >= MM) return;
    int b = r / TT;
    int t = r - b * TT;

    const float* lg = logits + (size_t)r * 12;
    const float* sc = scores + (size_t)r * 6;
    float x = fkp[(size_t)r * 2 + 0];
    float y = fkp[(size_t)r * 2 + 1];

    float best = 3.4e38f;
    int bi = 0;
    #pragma unroll
    for (int k = 0; k < KK; ++k) {
        float dx = lg[2 * k]     - x;
        float dy = lg[2 * k + 1] - y;
        float l = dx * dx + dy * dy;
        if (l < best) { best = l; bi = k; }   // first-min, matches argmin
    }
    selected[(size_t)r * 2 + 0] = lg[2 * bi];
    selected[(size_t)r * 2 + 1] = lg[2 * bi + 1];

    // faithful double softmax: ce = -log_softmax(softmax(s))[bi]
    float mx = sc[0];
    #pragma unroll
    for (int k = 1; k < KK; ++k) mx = fmaxf(mx, sc[k]);
    float e[KK], Z = 0.0f;
    #pragma unroll
    for (int k = 0; k < KK; ++k) { e[k] = expf(sc[k] - mx); Z += e[k]; }
    float inv = 1.0f / Z, mx2 = -3.4e38f;
    float sm[KK];
    #pragma unroll
    for (int k = 0; k < KK; ++k) { sm[k] = e[k] * inv; mx2 = fmaxf(mx2, sm[k]); }
    float Z2 = 0.0f;
    #pragma unroll
    for (int k = 0; k < KK; ++k) Z2 += expf(sm[k] - mx2);
    float ce = (mx2 + logf(Z2)) - sm[bi];

    minl_ws[(size_t)t * BB + b] = best;
    ce_ws[(size_t)t * BB + b]   = ce;
}

// ---------------------------------------------------------------------------
// Deterministic reduction over B for each t -> loss_per_kp[t]
// ---------------------------------------------------------------------------
__global__ void __launch_bounds__(256)
kp_reduce_kernel(const float* __restrict__ minl_ws,
                 const float* __restrict__ ce_ws,
                 float* __restrict__ loss_per_kp)   // [T]
{
    __shared__ float s1[256];
    __shared__ float s2[256];
    int t = blockIdx.x;
    float a = 0.0f, c = 0.0f;
    for (int b = threadIdx.x; b < BB; b += 256) {
        a += minl_ws[(size_t)t * BB + b];
        c += ce_ws[(size_t)t * BB + b];
    }
    s1[threadIdx.x] = a;
    s2[threadIdx.x] = c;
    __syncthreads();
    for (int off = 128; off > 0; off >>= 1) {
        if (threadIdx.x < off) {
            s1[threadIdx.x] += s1[threadIdx.x + off];
            s2[threadIdx.x] += s2[threadIdx.x + off];
        }
        __syncthreads();
    }
    if (threadIdx.x == 0) {
        float w = powf(0.1f, 4.0f - (float)t);   // weights[t] = 0.1^(4-t)
        loss_per_kp[t] = (s1[0] / (float)BB) * w + s2[0] / (float)BB;
    }
}

__global__ void kp_final_kernel(const float* __restrict__ loss_per_kp,
                                float* __restrict__ out0)
{
    if (threadIdx.x == 0 && blockIdx.x == 0) {
        float s = 0.0f;
        #pragma unroll
        for (int t = 0; t < TT; ++t) s += loss_per_kp[t];
        out0[0] = s / (float)TT;
    }
}

// ---------------------------------------------------------------------------
// Launch
// ---------------------------------------------------------------------------
extern "C" void kernel_launch(void* const* d_in, const int* in_sizes, int n_in,
                              void* d_out, int out_size, void* d_ws, size_t ws_size,
                              hipStream_t stream) {
    const float* hidden   = (const float*)d_in[0];
    const float* fkp      = (const float*)d_in[1];
    const float* ld_w1    = (const float*)d_in[2];
    const float* ld_b1    = (const float*)d_in[3];
    const float* ld_gamma = (const float*)d_in[4];
    const float* ld_beta  = (const float*)d_in[5];
    const float* ld_w2    = (const float*)d_in[6];
    const float* ld_b2    = (const float*)d_in[7];
    const float* sd_w1    = (const float*)d_in[8];
    const float* sd_b1    = (const float*)d_in[9];
    const float* sd_gamma = (const float*)d_in[10];
    const float* sd_beta  = (const float*)d_in[11];
    const float* sd_w2    = (const float*)d_in[12];
    const float* sd_b2    = (const float*)d_in[13];
    const int*   ctx      = (const int*)d_in[14];

    float* out     = (float*)d_out;
    float* logits  = out + OFF_LOGITS;
    float* scores  = out + OFF_SCORES;
    float* sel     = out + OFF_SEL;
    float* lpp     = out + OFF_LPP;

    char* ws = (char*)d_ws;
    __bf16* apack    = (__bf16*)(ws + WS_APACK_OFF);
    __bf16* bpack_ld = (__bf16*)(ws + WS_BPACK_LD_OFF);
    __bf16* bpack_sd = (__bf16*)(ws + WS_BPACK_SD_OFF);
    float*  minl_ws  = (float*)(ws + WS_RED_OFF);       // [T][B]
    float*  ce_ws    = minl_ws + (size_t)TT * BB;

    const size_t lds_bytes = (size_t)SMEM_FLOATS * sizeof(float); // 144 KB (WGP LDS = 320 KB)

    // One-shot precision split into fragment-ordered bf16 hi/lo buffers.
    kp_pack_a_kernel<<<(NTILE_M * NKC * 32 + 255) / 256, 256, 0, stream>>>(
        hidden, ctx, apack);
    kp_pack_w1_kernel<<<(64 * NKC * 32 + 255) / 256, 256, 0, stream>>>(
        ld_w1, bpack_ld);
    kp_pack_w1_kernel<<<(64 * NKC * 32 + 255) / 256, 256, 0, stream>>>(
        sd_w1, bpack_sd);

    kp_decoder_kernel<<<NTILE_M, 256, lds_bytes, stream>>>(
        hidden, apack, bpack_ld, ld_b1, ld_gamma, ld_beta, ld_w2, ld_b2,
        logits, ctx, 12);
    kp_decoder_kernel<<<NTILE_M, 256, lds_bytes, stream>>>(
        hidden, apack, bpack_sd, sd_b1, sd_gamma, sd_beta, sd_w2, sd_b2,
        scores, ctx, 6);

    kp_loss_kernel<<<(MM + 255) / 256, 256, 0, stream>>>(
        logits, scores, fkp, sel, minl_ws, ce_ws);
    kp_reduce_kernel<<<TT, 256, 0, stream>>>(minl_ws, ce_ws, lpp);
    kp_final_kernel<<<1, 32, 0, stream>>>(lpp, out);
}
